// ChessBoardAttention_90434831385118
// MI455X (gfx1250) — compile-verified, matrix operand order checked by reading, not run on hardware
//
#include <hip/hip_runtime.h>

// ---------------------------------------------------------------------------
// ChessBoard attention for MI455X (gfx1250, wave32, WMMA).
//   B=2, C=128, H=W=256, WS=8 -> 128 windows x 1024 tokens, d_qk=32, d_v=128.
// Pass 1: fused QKV projection (WMMA bf16, f32 acc) -> workspace (bf16).
// Pass 2: flash attention per window (WMMA bf16) + gamma*out + x epilogue.
// ---------------------------------------------------------------------------

typedef __attribute__((ext_vector_type(16))) __bf16 v16bf;
typedef __attribute__((ext_vector_type(8)))  __bf16 v8bf;
typedef __attribute__((ext_vector_type(8)))  float  v8f;

#define C_DIM   128
#define C4_DIM  32
#define H_DIM   256
#define W_DIM   256
#define NTOK    1024      // tokens per window (32x32)
#define NWIN    128       // B * WS * WS

// Fragment loader. CDNA5 16-bit A/B layout (ISA 7.12.2):
//   lane%16 selects M (A) or N (B); kh = lane/16.
//   elems 0..7  <- k = kc + kh*8 + i        (VGPR0..3)
//   elems 8..15 <- k = kc + 16 + kh*8 + i   (VGPR4..7)
// Both halves are 16B-contiguous -> two b128 loads.
__device__ __forceinline__ v16bf frag16(const __bf16* row, int kc, int kh) {
  v8bf lo = *(const v8bf*)(row + kc + kh * 8);
  v8bf hi = *(const v8bf*)(row + kc + 16 + kh * 8);
  v16bf r;
#pragma unroll
  for (int i = 0; i < 8; ++i) { r[i] = lo[i]; r[i + 8] = hi[i]; }
  return r;
}

// ---------------------------------------------------------------------------
// Pass 1: QKV projection.
// grid = NWIN * 32 blocks; each block: 32 tokens (one window row) x 192 dims.
// ---------------------------------------------------------------------------
#define XSTR 136   // LDS row stride (elems): 272B = 17*16B aligned, 4r mod 64 banks
#define WSTR 136

__global__ void __launch_bounds__(256)
qkv_proj_kernel(const float* __restrict__ x,
                const float* __restrict__ Wq, const float* __restrict__ bq,
                const float* __restrict__ Wk, const float* __restrict__ bk,
                const float* __restrict__ Wv, const float* __restrict__ bv,
                __bf16* __restrict__ Qws, __bf16* __restrict__ Kws,
                __bf16* __restrict__ Vws) {
  __shared__ __bf16 Xs[32 * XSTR];       // 32 tokens x 128 ch
  __shared__ __bf16 Wsh[192 * WSTR];     // fused [Wq;Wk;Wv] rows d=0..191
  __shared__ float  Bs[192];

  const int blk = blockIdx.x;            // 0..4095
  const int wd  = blk >> 5;              // window 0..127
  const int tb  = blk & 31;              // token-row block (window row i)
  const int b   = wd >> 6;
  const int ph  = wd & 63;
  const int sy  = ph >> 3, sx = ph & 7;
  const int tid = threadIdx.x;

  // ---- stage X tile: token j (0..31) = window col, h = tb*8+sy, w = j*8+sx
  {
    const int j  = tid & 31;
    const int cg = tid >> 5;             // 0..7, each covers 16 channels
    const int h  = tb * 8 + sy;
    const float* xp = x + (size_t)b * C_DIM * H_DIM * W_DIM
                        + (size_t)h * W_DIM + (j * 8 + sx);
#pragma unroll
    for (int cc = 0; cc < 16; ++cc) {
      const int c = cg * 16 + cc;
      Xs[j * XSTR + c] = (__bf16)xp[(size_t)c * H_DIM * W_DIM];
    }
  }
  // ---- stage fused weights (192 x 128), rows: [0,32)=Wq [32,64)=Wk [64,192)=Wv
  for (int idx = tid; idx < 192 * 128; idx += 256) {
    const int d = idx >> 7, c = idx & 127;
    float v;
    if (d < 32)       v = Wq[d * 128 + c];
    else if (d < 64)  v = Wk[(d - 32) * 128 + c];
    else              v = Wv[(d - 64) * 128 + c];
    Wsh[d * WSTR + c] = (__bf16)v;
  }
  if (tid < 192)
    Bs[tid] = (tid < 32) ? bq[tid] : (tid < 64 ? bk[tid - 32] : bv[tid - 64]);
  __syncthreads();

  const int wave = tid >> 5, lane = tid & 31;
  const int ln = lane & 15, kh = lane >> 4;
  const int mt = wave & 1;               // token tile (16 tokens)
  const int ng = wave >> 1;              // dim group: 48 dims each
  const __bf16* arow = Xs + (mt * 16 + ln) * XSTR;
  const int tokBase = tb * 32 + mt * 16;

#pragma unroll
  for (int nt = 0; nt < 3; ++nt) {
    const int dbase = ng * 48 + nt * 16;
    const __bf16* brow = Wsh + (dbase + ln) * WSTR;
    v8f acc = {};
#pragma unroll
    for (int kc = 0; kc < 4; ++kc) {
      v16bf a  = frag16(arow, kc * 32, kh);
      v16bf bm = frag16(brow, kc * 32, kh);
      acc = __builtin_amdgcn_wmma_f32_16x16x32_bf16(false, a, false, bm,
                                                    (short)0, acc, false, false);
    }
    const int   d    = dbase + ln;       // N = lane%16
    const float bias = Bs[d];
    const int   tok0 = tokBase + 8 * kh; // first of 8 consecutive token rows

    // Each 16-dim tile lies entirely in Q, K or V (tile-uniform branch).
    if (dbase < 32) {
      __bf16* dst = Qws + ((size_t)wd * NTOK + tok0) * 32 + d;
#pragma unroll
      for (int r = 0; r < 8; ++r) dst[r * 32] = (__bf16)(acc[r] + bias);
    } else if (dbase < 64) {
      __bf16* dst = Kws + ((size_t)wd * NTOK + tok0) * 32 + (d - 32);
#pragma unroll
      for (int r = 0; r < 8; ++r) dst[r * 32] = (__bf16)(acc[r] + bias);
    } else {
      // V transposed [win][dim][tok]: rows r are 8 consecutive tokens -> one
      // 16B vector store per lane.
      v8bf pk;
#pragma unroll
      for (int r = 0; r < 8; ++r) pk[r] = (__bf16)(acc[r] + bias);
      *(v8bf*)(Vws + ((size_t)wd * C_DIM + (d - 64)) * NTOK + tok0) = pk;
    }
  }
}

// ---------------------------------------------------------------------------
// Pass 2: flash attention. grid = NWIN * 8 blocks; block = 128 queries.
// ---------------------------------------------------------------------------
#define KSTR 40    // 32 dims + pad (80B, 16B-aligned)
#define VSTR 72    // 64 keys + pad (144B, 16B-aligned)
#define PSTR 72

__global__ void __launch_bounds__(256)
attn_kernel(const __bf16* __restrict__ Qws, const __bf16* __restrict__ Kws,
            const __bf16* __restrict__ Vws, const float* __restrict__ x,
            const float* __restrict__ gamma, float* __restrict__ out) {
  __shared__ __bf16 Ks[64 * KSTR];          // key block: [key][dim]
  __shared__ __bf16 Vs[128 * VSTR];         // value block: [vdim][key]
  __shared__ __bf16 Ps[8][16 * PSTR];       // per-wave P scratch: [m][key]

  const int blk  = blockIdx.x;              // 0..1023
  const int wd   = blk >> 3;                // window
  const int qb   = blk & 7;                 // query block of 128
  const int tid  = threadIdx.x;
  const int wave = tid >> 5, lane = tid & 31;
  const int ln = lane & 15, kh = lane >> 4;
  const int qBase = qb * 128 + wave * 16;

  // Q fragment (held in registers for the whole kernel)
  const __bf16* qrow = Qws + ((size_t)wd * NTOK + qBase + ln) * 32;
  const v16bf qf = frag16(qrow, 0, kh);

  float mrow[8], lrow[8];
  v8f o[8];
#pragma unroll
  for (int r = 0; r < 8; ++r) { mrow[r] = -1e30f; lrow[r] = 0.f; }
#pragma unroll
  for (int vt = 0; vt < 8; ++vt) o[vt] = (v8f){};

  for (int kb = 0; kb < 16; ++kb) {
    // ---- stage K block: 64 keys x 32 dims (one b128 per thread)
    {
      const int key = tid >> 2, part = tid & 3;
      *(v8bf*)(Ks + key * KSTR + part * 8) =
          *(const v8bf*)(Kws + ((size_t)wd * NTOK + kb * 64 + key) * 32 + part * 8);
    }
    // ---- stage V block: 128 dims x 64 keys (4x b128 per thread)
    {
      const int dim = tid >> 1, half = tid & 1;
      const __bf16* src = Vws + ((size_t)wd * C_DIM + dim) * NTOK + kb * 64 + half * 32;
      __bf16* dst = Vs + dim * VSTR + half * 32;
#pragma unroll
      for (int q4 = 0; q4 < 4; ++q4)
        *(v8bf*)(dst + q4 * 8) = *(const v8bf*)(src + q4 * 8);
    }
    if (kb < 15) {  // prefetch next key block (global_prefetch_b8)
      __builtin_prefetch(Kws + ((size_t)wd * NTOK + (kb + 1) * 64 + (tid >> 2)) * 32, 0, 1);
    }
    __syncthreads();

    // ---- S = Q * K^T  (d=32 -> exactly one WMMA per 16-key tile)
    v8f s[4];
#pragma unroll
    for (int nt = 0; nt < 4; ++nt) {
      const __bf16* krow = Ks + (nt * 16 + ln) * KSTR;
      v16bf kf = frag16(krow, 0, kh);
      v8f z = {};
      s[nt] = __builtin_amdgcn_wmma_f32_16x16x32_bf16(false, qf, false, kf,
                                                      (short)0, z, false, false);
    }

    // ---- online softmax; row m = r + 8*kh lives across 16 lanes of one half
#pragma unroll
    for (int r = 0; r < 8; ++r) {
      float v = fmaxf(fmaxf(s[0][r], s[1][r]), fmaxf(s[2][r], s[3][r]));
#pragma unroll
      for (int off = 1; off < 16; off <<= 1) v = fmaxf(v, __shfl_xor(v, off, 32));
      const float mnew = fmaxf(mrow[r], v);
      const float corr = __expf(mrow[r] - mnew);
      mrow[r] = mnew;
      float rsum = 0.f;
#pragma unroll
      for (int nt = 0; nt < 4; ++nt) {
        const float p = __expf(s[nt][r] - mnew);
        s[nt][r] = p;
        rsum += p;
      }
#pragma unroll
      for (int off = 1; off < 16; off <<= 1) rsum += __shfl_xor(rsum, off, 32);
      lrow[r] = lrow[r] * corr + rsum;
#pragma unroll
      for (int vt = 0; vt < 8; ++vt) o[vt][r] *= corr;
    }

    // ---- P: C/D layout -> A layout via per-wave LDS round trip
    __bf16* pbase = &Ps[wave][0];
#pragma unroll
    for (int nt = 0; nt < 4; ++nt)
#pragma unroll
      for (int r = 0; r < 8; ++r)
        pbase[(r + 8 * kh) * PSTR + nt * 16 + ln] = (__bf16)s[nt][r];

    const __bf16* prow = pbase + ln * PSTR;
    v16bf p0 = frag16(prow, 0, kh);
    v16bf p1 = frag16(prow, 32, kh);

    // ---- O += P * V  (8 vtiles x 2 K-chunks)
#pragma unroll
    for (int vt = 0; vt < 8; ++vt) {
      const __bf16* vrow = Vs + (vt * 16 + ln) * VSTR;
      v16bf v0 = frag16(vrow, 0, kh);
      v16bf v1 = frag16(vrow, 32, kh);
      o[vt] = __builtin_amdgcn_wmma_f32_16x16x32_bf16(false, p0, false, v0,
                                                      (short)0, o[vt], false, false);
      o[vt] = __builtin_amdgcn_wmma_f32_16x16x32_bf16(false, p1, false, v1,
                                                      (short)0, o[vt], false, false);
    }
    __syncthreads();
  }

  // ---- epilogue: out = gamma * (O / l) + x, scattered to chessboard pixels
  const int b  = wd >> 6;
  const int ph = wd & 63;
  const int sy = ph >> 3, sx = ph & 7;
  const float g = gamma[0];
  float invl[8];
#pragma unroll
  for (int r = 0; r < 8; ++r) invl[r] = 1.0f / lrow[r];
#pragma unroll
  for (int vt = 0; vt < 8; ++vt) {
    const int cdim = vt * 16 + ln;
#pragma unroll
    for (int r = 0; r < 8; ++r) {
      const int tok = qBase + r + 8 * kh;
      const int i = tok >> 5, j = tok & 31;
      const int hh = i * 8 + sy, ww = j * 8 + sx;
      const size_t idx = (((size_t)b * C_DIM + cdim) * H_DIM + hh) * W_DIM + ww;
      const float val = o[vt][r] * invl[r];
      out[idx] = g * val + x[idx];
    }
  }
}

// ---------------------------------------------------------------------------
extern "C" void kernel_launch(void* const* d_in, const int* in_sizes, int n_in,
                              void* d_out, int out_size, void* d_ws, size_t ws_size,
                              hipStream_t stream) {
  const float* x     = (const float*)d_in[0];
  const float* Wq    = (const float*)d_in[1];
  const float* bq    = (const float*)d_in[2];
  const float* Wk    = (const float*)d_in[3];
  const float* bk    = (const float*)d_in[4];
  const float* Wv    = (const float*)d_in[5];
  const float* bv    = (const float*)d_in[6];
  const float* gamma = (const float*)d_in[7];
  float* out = (float*)d_out;

  // Workspace layout (bf16): Q[128][1024][32], K[128][1024][32], V[128][128][1024]
  __bf16* Qws = (__bf16*)d_ws;
  __bf16* Kws = Qws + (size_t)NWIN * NTOK * C4_DIM;
  __bf16* Vws = Kws + (size_t)NWIN * NTOK * C4_DIM;

  qkv_proj_kernel<<<NWIN * 32, 256, 0, stream>>>(x, Wq, bq, Wk, bk, Wv, bv,
                                                 Qws, Kws, Vws);
  attn_kernel<<<NWIN * 8, 256, 0, stream>>>(Qws, Kws, Vws, x, gamma, out);
}
